// GAT_2645699854496
// MI455X (gfx1250) — matrix operand order repaired and checked by reading, not compiled, not run on hardware
//
#include <hip/hip_runtime.h>
#include <hip/hip_bf16.h>
#include <cstring>
#include <cstddef>

// ---------------------------------------------------------------------------
// Types for CDNA5 WMMA
// ---------------------------------------------------------------------------
typedef __attribute__((ext_vector_type(16))) __bf16        v16bf;
typedef __attribute__((ext_vector_type(8)))  float         v8f;
typedef __attribute__((ext_vector_type(8)))  unsigned int  v8u;
typedef __attribute__((ext_vector_type(4)))  unsigned int  v4u;

union Frag16 { v16bf v; v8u u; };

// float -> bf16 bits, round-to-nearest-even
__device__ __forceinline__ unsigned short f2bf_rne(float f) {
    unsigned int u = __float_as_uint(f);
    u += 0x7FFFu + ((u >> 16) & 1u);
    return (unsigned short)(u >> 16);
}
__device__ __forceinline__ float bf2f(unsigned short h) {
    return __uint_as_float(((unsigned int)h) << 16);
}

// order-preserving float<->uint encoding (for atomic float max via atomicMax u32)
__device__ __forceinline__ unsigned int fenc(float f) {
    unsigned int u = __float_as_uint(f);
    return (u & 0x80000000u) ? ~u : (u | 0x80000000u);
}
__device__ __forceinline__ float fdec(unsigned int e) {
    unsigned int u = (e & 0x80000000u) ? (e & 0x7FFFFFFFu) : ~e;
    return __uint_as_float(u);
}

// ---------------------------------------------------------------------------
// GEMM: C[M,NCOLS] = A[M,KDIM] * B[KDIM,NCOLS]  (fp32 in/out)
// bf16x2 split (hi/lo) with v_wmma_f32_16x16x32_bf16, fp32 accumulate.
// Block: 256 threads = 8 waves; each wave owns a 16-row strip of a 128-row tile.
// ---------------------------------------------------------------------------
template<int KDIM, int NCOLS, int NPAD>
__global__ __launch_bounds__(256) void gemm_bf16x2_kernel(
    const float* __restrict__ A, const float* __restrict__ B,
    float* __restrict__ C, int M)
{
    constexpr int NT = NPAD / 16;
    __shared__ __align__(32) unsigned short sAhi[128 * 32];
    __shared__ __align__(32) unsigned short sAlo[128 * 32];
    __shared__ __align__(32) unsigned short sBhi[NPAD * 32];
    __shared__ __align__(32) unsigned short sBlo[NPAD * 32];

    const int tid  = threadIdx.x;
    const int lane = tid & 31;
    const int wave = tid >> 5;
    const int rowBase = blockIdx.x * 128;
    const bool fullTile = (rowBase + 128 <= M);

    v8f acc[NT];
    #pragma unroll
    for (int t = 0; t < NT; ++t)
        #pragma unroll
        for (int j = 0; j < 8; ++j) acc[t][j] = 0.0f;

    for (int kc = 0; kc < KDIM; kc += 32) {
        // ---- stage A tile: 128x32 fp32 -> bf16 hi/lo in LDS (float4 loads) ----
        for (int f = tid; f < 128 * 8; f += 256) {
            const int r  = f >> 3;
            const int c4 = f & 7;
            const int grow = rowBase + r;
            float4 v;
            if (fullTile || grow < M)
                v = *(const float4*)&A[(size_t)grow * KDIM + kc + c4 * 4];
            else
                v = make_float4(0.f, 0.f, 0.f, 0.f);
            const float vv[4] = { v.x, v.y, v.z, v.w };
            #pragma unroll
            for (int j = 0; j < 4; ++j) {
                const unsigned short hi = f2bf_rne(vv[j]);
                const float lo = vv[j] - bf2f(hi);
                sAhi[r * 32 + c4 * 4 + j] = hi;
                sAlo[r * 32 + c4 * 4 + j] = f2bf_rne(lo);
            }
        }
        // ---- stage B tile transposed: sBt[n][k] = B[kc+k][n], zero-pad n>=NCOLS ----
        for (int idx = tid; idx < NPAD * 32; idx += 256) {
            const int n = idx >> 5;
            const int k = idx & 31;
            const float bv = (n < NCOLS) ? B[(size_t)(kc + k) * NCOLS + n] : 0.0f;
            const unsigned short hi = f2bf_rne(bv);
            sBhi[idx] = hi;
            sBlo[idx] = f2bf_rne(bv - bf2f(hi));
        }
        __syncthreads();

        // ---- A fragments (hi/lo) lane half h holds k = 16h..16h+15, as B does ----
        Frag16 ah, al;
        {
            const int m    = wave * 16 + (lane & 15);
            const int koff = (lane >> 4) * 16;
            const v4u a0 = *(const v4u*)(sAhi + m * 32 + koff);
            const v4u a1 = *(const v4u*)(sAhi + m * 32 + koff + 8);
            const v4u l0 = *(const v4u*)(sAlo + m * 32 + koff);
            const v4u l1 = *(const v4u*)(sAlo + m * 32 + koff + 8);
            #pragma unroll
            for (int j = 0; j < 4; ++j) {
                ah.u[j] = a0[j]; ah.u[4 + j] = a1[j];
                al.u[j] = l0[j]; al.u[4 + j] = l1[j];
            }
        }
        // ---- per n-tile: B fragments + 3 WMMAs (hi*hi + hi*lo + lo*hi) ----
        #pragma unroll
        for (int t = 0; t < NT; ++t) {
            const int n  = t * 16 + (lane & 15);
            const int kb = (lane >> 4) * 16;
            Frag16 bh, bl;
            bh.u = *(const v8u*)(sBhi + n * 32 + kb);
            bl.u = *(const v8u*)(sBlo + n * 32 + kb);
            acc[t] = __builtin_amdgcn_wmma_f32_16x16x32_bf16(
                false, ah.v, false, bh.v, (short)0, acc[t], false, false);
            acc[t] = __builtin_amdgcn_wmma_f32_16x16x32_bf16(
                false, ah.v, false, bl.v, (short)0, acc[t], false, false);
            acc[t] = __builtin_amdgcn_wmma_f32_16x16x32_bf16(
                false, al.v, false, bh.v, (short)0, acc[t], false, false);
        }
        __syncthreads();
    }

    // ---- epilogue: C/D layout row = (lane>>4)*8 + r, col = lane&15 ----
    const int mloc = wave * 16 + (lane >> 4) * 8;
    const int col0 = lane & 15;
    if (fullTile) {
        // fast path: no row guards -> straight-line global stores
        #pragma unroll
        for (int t = 0; t < NT; ++t) {
            const int col = t * 16 + col0;
            if (NPAD == NCOLS || col < NCOLS) {
                #pragma unroll
                for (int r = 0; r < 8; ++r)
                    C[(size_t)(rowBase + mloc + r) * NCOLS + col] = acc[t][r];
            }
        }
    } else {
        #pragma unroll
        for (int t = 0; t < NT; ++t) {
            const int col = t * 16 + col0;
            #pragma unroll
            for (int r = 0; r < 8; ++r) {
                const int row = rowBase + mloc + r;
                if (row < M && (NPAD == NCOLS || col < NCOLS))
                    C[(size_t)row * NCOLS + col] = acc[t][r];
            }
        }
    }
}

// ---------------------------------------------------------------------------
// attention dot products: as[n,h] = sum_c h[n,h,c]*att_s[h,c] (same for ad)
// ---------------------------------------------------------------------------
template<int H, int CC>
__global__ void att_kernel(const float* __restrict__ h,
                           const float* __restrict__ att_s,
                           const float* __restrict__ att_d,
                           float* __restrict__ as_, float* __restrict__ ad_,
                           int N)
{
    const int i = blockIdx.x * blockDim.x + threadIdx.x;
    if (i >= N * H) return;
    const int hh = i % H;                       // compile-time H -> mask/shift
    const float* hp = &h[(size_t)i * CC];       // [n*H + hh] * CC
    float ss = 0.f, dd = 0.f;
    #pragma unroll
    for (int c = 0; c < CC; ++c) {
        const float v = hp[c];
        ss += v * att_s[hh * CC + c];
        dd += v * att_d[hh * CC + c];
    }
    as_[i] = ss; ad_[i] = dd;
}

__global__ void fill_u32_kernel(unsigned int* __restrict__ p, unsigned int v, int n)
{
    const int i = blockIdx.x * blockDim.x + threadIdx.x;
    if (i < n) p[i] = v;
}

// ---------------------------------------------------------------------------
// Edge passes. Edge list augmented with self-loops: e in [E, E+N) -> s=d=e-E.
// ---------------------------------------------------------------------------
template<int H>
__device__ __forceinline__ float edge_e(const float* as_, const float* ad_,
                                        int s, int d, int h)
{
    float v = as_[s * H + h] + ad_[d * H + h];
    return v >= 0.f ? v : 0.2f * v;   // leaky_relu(0.2)
}

template<int H>
__global__ void edge_max_kernel(const int* __restrict__ src, const int* __restrict__ dst,
                                int E, int N,
                                const float* __restrict__ as_, const float* __restrict__ ad_,
                                unsigned int* __restrict__ emax)
{
    const int i = blockIdx.x * blockDim.x + threadIdx.x;
    if (i >= (E + N) * H) return;
    const int e = i / H, h = i % H;
    int s, d;
    if (e < E) { s = src[e]; d = dst[e]; } else { s = d = e - E; }
    const float v = edge_e<H>(as_, ad_, s, d, h);
    atomicMax(&emax[d * H + h], fenc(v));
}

template<int H>
__global__ void edge_sum_kernel(const int* __restrict__ src, const int* __restrict__ dst,
                                int E, int N,
                                const float* __restrict__ as_, const float* __restrict__ ad_,
                                const unsigned int* __restrict__ emax,
                                float* __restrict__ denom)
{
    const int i = blockIdx.x * blockDim.x + threadIdx.x;
    if (i >= (E + N) * H) return;
    const int e = i / H, h = i % H;
    int s, d;
    if (e < E) { s = src[e]; d = dst[e]; } else { s = d = e - E; }
    const float v = edge_e<H>(as_, ad_, s, d, h);
    atomicAdd(&denom[d * H + h], __expf(v - fdec(emax[d * H + h])));
}

template<int H, int CC>
__global__ void edge_agg_kernel(const int* __restrict__ src, const int* __restrict__ dst,
                                int E, int N,
                                const float* __restrict__ as_, const float* __restrict__ ad_,
                                const unsigned int* __restrict__ emax,
                                const float* __restrict__ denom,
                                const float* __restrict__ hfeat,
                                float* __restrict__ agg)
{
    const int i = blockIdx.x * blockDim.x + threadIdx.x;
    if (i >= (E + N) * H) return;
    const int e = i / H, h = i % H;
    int s, d;
    if (e < E) { s = src[e]; d = dst[e]; } else { s = d = e - E; }
    const float v  = edge_e<H>(as_, ad_, s, d, h);
    const float ex = __expf(v - fdec(emax[d * H + h]));
    const float alpha = ex / (denom[d * H + h] + 1e-16f);
    const float* hs = &hfeat[((size_t)s * H + h) * CC];
    float* ag       = &agg  [((size_t)d * H + h) * CC];
    // vectorized gather of h[src] (CC % 4 == 0), then scatter-atomics
    #pragma unroll
    for (int c4 = 0; c4 < CC / 4; ++c4) {
        const float4 hv = *(const float4*)&hs[c4 * 4];
        atomicAdd(&ag[c4 * 4 + 0], alpha * hv.x);
        atomicAdd(&ag[c4 * 4 + 1], alpha * hv.y);
        atomicAdd(&ag[c4 * 4 + 2], alpha * hv.z);
        atomicAdd(&ag[c4 * 4 + 3], alpha * hv.w);
    }
}

template<int F>
__global__ void bias_relu_kernel(const float* __restrict__ agg, const float* __restrict__ b,
                                 float* __restrict__ z, int total)
{
    const int i = blockIdx.x * blockDim.x + threadIdx.x;
    if (i >= total) return;
    const float v = agg[i] + b[i % F];          // F = 64 -> mask
    z[i] = v > 0.f ? v : 0.f;
}

// y = agg2 + b2 ; out = log_softmax(y) over 40 classes
__global__ void final_lsm_kernel(const float* __restrict__ agg2, const float* __restrict__ b2,
                                 float* __restrict__ out, int N)
{
    const int n = blockIdx.x * blockDim.x + threadIdx.x;
    if (n >= N) return;
    float y[40];
    float mx = -3.0e38f;
    #pragma unroll
    for (int c = 0; c < 40; ++c) {
        y[c] = agg2[(size_t)n * 40 + c] + b2[c];
        mx = fmaxf(mx, y[c]);
    }
    float sum = 0.f;
    #pragma unroll
    for (int c = 0; c < 40; ++c) sum += __expf(y[c] - mx);
    const float lse = __logf(sum) + mx;
    #pragma unroll
    for (int c = 0; c < 40; ++c) out[(size_t)n * 40 + c] = y[c] - lse;
}

// ---------------------------------------------------------------------------
// Launcher
// ---------------------------------------------------------------------------
extern "C" void kernel_launch(void* const* d_in, const int* in_sizes, int n_in,
                              void* d_out, int out_size, void* d_ws, size_t ws_size,
                              hipStream_t stream)
{
    const float* x    = (const float*)d_in[0];
    const int*   ei   = (const int*)  d_in[1];   // [2,E] flat: row0=src, row1=dst
    const float* W1   = (const float*)d_in[2];
    const float* as1w = (const float*)d_in[3];
    const float* ad1w = (const float*)d_in[4];
    const float* b1   = (const float*)d_in[5];
    const float* W2   = (const float*)d_in[6];
    const float* as2w = (const float*)d_in[7];
    const float* ad2w = (const float*)d_in[8];
    const float* b2   = (const float*)d_in[9];

    const int N = in_sizes[0] / 512;         // 50000
    const int E = in_sizes[1] / 2;           // 1600000
    const int* src = ei;
    const int* dst = ei + E;

    // workspace carve-up
    char* ws = (char*)d_ws;
    auto take = [&](size_t bytes) { char* p = ws; ws += (bytes + 255) & ~(size_t)255; return p; };
    float*        h1    = (float*)       take((size_t)N * 64 * 4);
    float*        z     = (float*)       take((size_t)N * 64 * 4);
    float*        agg1  = (float*)       take((size_t)N * 64 * 4);
    float*        as1   = (float*)       take((size_t)N * 8 * 4);
    float*        ad1   = (float*)       take((size_t)N * 8 * 4);
    unsigned int* emax1 = (unsigned int*)take((size_t)N * 8 * 4);
    float*        den1  = (float*)       take((size_t)N * 8 * 4);
    float*        h2    = (float*)       take((size_t)N * 40 * 4);
    float*        agg2  = (float*)       take((size_t)N * 40 * 4);
    float*        as2   = (float*)       take((size_t)N * 4);
    float*        ad2   = (float*)       take((size_t)N * 4);
    unsigned int* emax2 = (unsigned int*)take((size_t)N * 4);
    float*        den2  = (float*)       take((size_t)N * 4);

    // encoded -inf sentinel for float-max-as-uint
    float negBig = -3.0e38f; unsigned int ub; memcpy(&ub, &negBig, 4);
    const unsigned int encNeg = ~ub;   // negBig has sign bit set -> enc = ~u

    const int TB = 256;
    auto blk = [&](long long n) { return (int)((n + TB - 1) / TB); };

    // ---- init accumulators ----
    fill_u32_kernel<<<blk(N * 64), TB, 0, stream>>>((unsigned int*)agg1, 0u, N * 64);
    fill_u32_kernel<<<blk(N * 8),  TB, 0, stream>>>(emax1, encNeg, N * 8);
    fill_u32_kernel<<<blk(N * 8),  TB, 0, stream>>>((unsigned int*)den1, 0u, N * 8);
    fill_u32_kernel<<<blk(N * 40), TB, 0, stream>>>((unsigned int*)agg2, 0u, N * 40);
    fill_u32_kernel<<<blk(N),      TB, 0, stream>>>(emax2, encNeg, N);
    fill_u32_kernel<<<blk(N),      TB, 0, stream>>>((unsigned int*)den2, 0u, N);

    // ---- layer 1: GAT(512 -> 8 heads x 8) ----
    gemm_bf16x2_kernel<512, 64, 64><<<(N + 127) / 128, 256, 0, stream>>>(x, W1, h1, N);
    att_kernel<8, 8><<<blk(N * 8), TB, 0, stream>>>(h1, as1w, ad1w, as1, ad1, N);

    const int EH1 = (E + N) * 8;
    edge_max_kernel<8><<<blk(EH1), TB, 0, stream>>>(src, dst, E, N, as1, ad1, emax1);
    edge_sum_kernel<8><<<blk(EH1), TB, 0, stream>>>(src, dst, E, N, as1, ad1, emax1, den1);
    edge_agg_kernel<8, 8><<<blk(EH1), TB, 0, stream>>>(src, dst, E, N, as1, ad1, emax1, den1, h1, agg1);

    bias_relu_kernel<64><<<blk(N * 64), TB, 0, stream>>>(agg1, b1, z, N * 64);

    // ---- layer 2: GAT(64 -> 1 head x 40) ----
    gemm_bf16x2_kernel<64, 40, 48><<<(N + 127) / 128, 256, 0, stream>>>(z, W2, h2, N);
    att_kernel<1, 40><<<blk(N), TB, 0, stream>>>(h2, as2w, ad2w, as2, ad2, N);

    const int EH2 = (E + N);
    edge_max_kernel<1><<<blk(EH2), TB, 0, stream>>>(src, dst, E, N, as2, ad2, emax2);
    edge_sum_kernel<1><<<blk(EH2), TB, 0, stream>>>(src, dst, E, N, as2, ad2, emax2, den2);
    edge_agg_kernel<1, 40><<<blk(EH2), TB, 0, stream>>>(src, dst, E, N, as2, ad2, emax2, den2, h2, agg2);

    final_lsm_kernel<<<blk(N), TB, 0, stream>>>(agg2, b2, (float*)d_out, N);
}
